// ImprovedProjectionFusionModel_37452114821172
// MI455X (gfx1250) — matrix operand order repaired and checked by reading, not compiled, not run on hardware
//
#include <hip/hip_runtime.h>
#include <hip/hip_bf16.h>
#include <math.h>

// ---------------------------------------------------------------------------
// CDNA5 (gfx1250) WMMA types
// ---------------------------------------------------------------------------
typedef __bf16 bf16_t;
typedef bf16_t v16bf __attribute__((ext_vector_type(16)));
typedef float  v8f   __attribute__((ext_vector_type(8)));

#define V_TOT 150000
#define CAMS  6
#define DINO  384
#define HFE   64
#define WFE   112
#define HID   128
#define NCLS  32
#define HBEV  200
#define WBEV  200
#define NCELL (HBEV*WBEV)
#define XMINF (-50.0f)
#define YMINF (-50.0f)
#define RESF  (0.5f)

__device__ __forceinline__ v8f wmma_bf16(v16bf a, v16bf b, v8f c) {
  return __builtin_amdgcn_wmma_f32_16x16x32_bf16(false, a, false, b, (short)0, c,
                                                 false, false);
}

// --- WMMA 16-bit operand swizzle (ISA 7.12.2) ----------------------------
// A 16x32 per K-chunk: lane l (half=l>>4, row=l&15) needs, in element order
// i=0..15, the K values  (i<8?0:16) + half*8 + (i&7).
// We store operands so each lane's 16 bf16 are contiguous (32B):
//   addr16 = ((row*nchunk + chunk)*2 + half)*16 + i
// Inverse map for writers holding channel c within the row:
__device__ __forceinline__ int swz_idx(int row, int nchunk, int c) {
  int chunk = c >> 5, sub = c & 31;
  int half  = (sub >> 3) & 1;
  int i     = (sub & 7) | ((sub & 16) ? 8 : 0);
  return ((row * nchunk + chunk) * 2 + half) * 16 + i;
}
// B 32x16 per K-chunk: lane l needs column n0+(l&15), K = (i<8?0:16) + half*8 + (i&7), as A.
// Packed weight layout: pk[((chunk*N + n)*2 + half)*16 + i]  (32B per lane).

// ===========================================================================
// Weight packing: fp32 row-major [K x N] (zero-padded to Kpad) -> bf16 B-frag
// ===========================================================================
__global__ void pack_weight_kernel(const float* __restrict__ W, bf16_t* __restrict__ out,
                                   int K, int N, int total) {
  int e = blockIdx.x * 256 + threadIdx.x;
  if (e >= total) return;
  int i     = e & 15;
  int half  = (e >> 4) & 1;
  int n     = (e >> 5) % N;
  int chunk = e / (32 * N);
  int kk    = chunk * 32 + half * 8 + (i & 7) + ((i & 8) ? 16 : 0);
  float v   = (kk < K) ? W[(size_t)kk * N + n] : 0.f;
  out[e] = (bf16_t)v;
}

// conv OIHW [128,128,3,3] -> implicit-GEMM B with K = tap*128 + ci  (36 chunks)
__global__ void pack_conv_kernel(const float* __restrict__ w, bf16_t* __restrict__ out) {
  int e = blockIdx.x * 256 + threadIdx.x;          // total 36*128*32 = 147456
  int i     = e & 15;
  int half  = (e >> 4) & 1;
  int n     = (e >> 5) & (HID - 1);
  int chunk = e >> 12;                             // / (32*128)
  int kk    = chunk * 32 + half * 8 + (i & 7) + ((i & 8) ? 16 : 0);
  int ci    = kk & (HID - 1);
  int tap   = kk >> 7;
  out[e] = (bf16_t)w[(((size_t)n * HID + ci) * 3 + tap / 3) * 3 + (tap % 3)];
}

// ===========================================================================
// Kernel A: projection + LN + img/lidar/gate GEMMs (WMMA bf16) + fuse + scatter
// block = 256 threads (8 waves), one block per 16 points (150000/16 = 9375)
// ===========================================================================
__global__ void __launch_bounds__(256)
fuse_point_kernel(const float* __restrict__ pf,        // V x 16
                  const float* __restrict__ imgfeat,   // 6 x 384 x 64 x 112
                  const float* __restrict__ coords,    // V x 3
                  const float* __restrict__ calib,     // 6 x 3 x 4
                  const float* __restrict__ lidar_ln_g, const float* __restrict__ lidar_ln_b,
                  const bf16_t* __restrict__ lid_pk,   const float* __restrict__ lidar_b,
                  const float* __restrict__ img_ln_g,  const float* __restrict__ img_ln_b,
                  const bf16_t* __restrict__ img_pk,   const float* __restrict__ img_b,
                  const bf16_t* __restrict__ gate_pk,  const float* __restrict__ gate_b1,
                  const float* __restrict__ gate_w2,   const float* __restrict__ gate_b2,
                  const int* __restrict__ himgp, const int* __restrict__ wimgp,
                  float* __restrict__ fusedG,
                  float* __restrict__ bev_img, float* __restrict__ bev_lid,
                  float* __restrict__ cnt)
{
  __shared__ float  sAcc[16 * DINO];                               // 24 KB; later imgS/lidS
  __shared__ __attribute__((aligned(32))) bf16_t sBf[16 * DINO];   // 12 KB swizzled A
  __shared__ float  sH  [16 * HID];                                // 8 KB gate hidden
  __shared__ __attribute__((aligned(32))) bf16_t sL [16 * 32];     // 1 KB lidar A (swz)
  __shared__ float  sRed[2][16][16];
  __shared__ float  sXs[16], sYs[16];
  __shared__ int    sMsk[16];
  __shared__ float  sCnt[16], sMean[16], sInv[16], sGate[16];
  __shared__ int    sBevIdx[16], sBevM[16];

  const int t    = threadIdx.x;
  const int lane = t & 31;
  const int wid  = t >> 5;
  const int half = lane >> 4;
  const int mrow = lane & 15;
  const int v0   = blockIdx.x * 16;
  const int p    = t >> 4;     // point within tile
  const int cg   = t & 15;     // channel group

  const float wimg = (float)(*wimgp);
  const float himg = (float)(*himgp);

  for (int j = t; j < 16 * DINO; j += 256) sAcc[j] = 0.0f;
  if (t < 16) sCnt[t] = 0.0f;

  float cx = 0.f, cy = 0.f, cz = 0.f;
  if (t < 16) {
    cx = coords[(size_t)(v0 + t) * 3 + 0];
    cy = coords[(size_t)(v0 + t) * 3 + 1];
    cz = coords[(size_t)(v0 + t) * 3 + 2];
  }
  __syncthreads();

  // ---- projection + bilinear accumulate over cameras ----
  for (int cam = 0; cam < CAMS; ++cam) {
    if (t < 16) {
      const float* M = calib + cam * 12;
      float p0 = M[0] * cx + M[1] * cy + M[2]  * cz + M[3];
      float p1 = M[4] * cx + M[5] * cy + M[6]  * cz + M[7];
      float p2 = M[8] * cx + M[9] * cy + M[10] * cz + M[11];
      float wm = fmaxf(p2, 1e-6f);
      float u  = p0 / wm, vv = p1 / wm;
      int m = (p2 > 0.f) && (u >= 0.f) && (u < wimg) && (vv >= 0.f) && (vv < himg);
      sXs[t]  = u  * (float)(WFE - 1) / wimg;
      sYs[t]  = vv * (float)(HFE - 1) / himg;
      sMsk[t] = m;
      sCnt[t] += (float)m;
    }
    __syncthreads();
    if (sMsk[p]) {
      float xs = sXs[p], ys = sYs[p];
      float x0f = floorf(xs), y0f = floorf(ys);
      int   x0i = (int)x0f,   y0i = (int)y0f;
      float wx = xs - x0f, wy = ys - y0f;
      bool vx0 = (x0i >= 0)     && (x0i < WFE);
      bool vx1 = (x0i + 1 >= 0) && (x0i + 1 < WFE);
      bool vy0 = (y0i >= 0)     && (y0i < HFE);
      bool vy1 = (y0i + 1 >= 0) && (y0i + 1 < HFE);
      float w00 = (vx0 && vy0) ? (1.f - wx) * (1.f - wy) : 0.f;
      float w10 = (vx1 && vy0) ? wx * (1.f - wy)         : 0.f;
      float w01 = (vx0 && vy1) ? (1.f - wx) * wy         : 0.f;
      float w11 = (vx1 && vy1) ? wx * wy                 : 0.f;
      int x0c = min(max(x0i, 0), WFE - 1), x1c = min(max(x0i + 1, 0), WFE - 1);
      int y0c = min(max(y0i, 0), HFE - 1), y1c = min(max(y0i + 1, 0), HFE - 1);
      const float* fb = imgfeat + (size_t)cam * DINO * HFE * WFE;
      int i00 = y0c * WFE + x0c, i10 = y0c * WFE + x1c;
      int i01 = y1c * WFE + x0c, i11 = y1c * WFE + x1c;
      for (int j = 0; j < 24; ++j) {
        int c = cg + 16 * j;
        const float* fc = fb + (size_t)c * (HFE * WFE);
        sAcc[p * DINO + c] += w00 * fc[i00] + w10 * fc[i10] + w01 * fc[i01] + w11 * fc[i11];
      }
    }
    __syncthreads();
  }

  // ---- vis = acc / max(cnt,1) ; LayerNorm over 384 ; -> swizzled bf16 ----
  {
    float d = sCnt[p]; d = (d == 0.f) ? 1.f : d;
    float inv = 1.f / d;
    float ps = 0.f, pq = 0.f;
    for (int j = 0; j < 24; ++j) {
      int c = cg + 16 * j;
      float x = sAcc[p * DINO + c] * inv;
      sAcc[p * DINO + c] = x;
      ps += x; pq += x * x;
    }
    sRed[0][p][cg] = ps; sRed[1][p][cg] = pq;
  }
  __syncthreads();
  if (cg == 0) {
    float s = 0.f, q = 0.f;
    for (int k = 0; k < 16; ++k) { s += sRed[0][p][k]; q += sRed[1][p][k]; }
    float mean = s * (1.0f / DINO);
    float var  = fmaxf(q * (1.0f / DINO) - mean * mean, 0.f);
    sMean[p] = mean; sInv[p] = rsqrtf(var + 1e-5f);
  }
  __syncthreads();
  for (int j = 0; j < 24; ++j) {
    int c = cg + 16 * j;
    float x = (sAcc[p * DINO + c] - sMean[p]) * sInv[p] * img_ln_g[c] + img_ln_b[c];
    sBf[swz_idx(p, 12, c)] = (bf16_t)x;
  }
  // ---- lidar LN (16 features) -> swizzled, zero-padded K=32 ----
  if (t < 16) {
    float vals[16]; float s = 0.f, q = 0.f;
    for (int k = 0; k < 16; ++k) {
      float x = pf[(size_t)(v0 + t) * 16 + k];
      vals[k] = x; s += x; q += x * x;
    }
    float mean = s * (1.f / 16.f);
    float var  = fmaxf(q * (1.f / 16.f) - mean * mean, 0.f);
    float inv  = rsqrtf(var + 1e-5f);
    for (int k = 0; k < 16; ++k)
      sL[swz_idx(t, 1, k)] = (bf16_t)((vals[k] - mean) * inv * lidar_ln_g[k] + lidar_ln_b[k]);
    for (int k = 16; k < 32; ++k) sL[swz_idx(t, 1, k)] = (bf16_t)0.f;
  }
  __syncthreads();   // sAcc free from here

  const int ncol = wid * 16 + mrow;   // this wave's output column (0..127)

  // ---- img GEMM: (16x384)@(384x128): 12 WMMA, 2 ds_b128 + 2 glb_b128 per frag ----
  {
    v8f acc = {0.f,0.f,0.f,0.f,0.f,0.f,0.f,0.f};
    for (int ch = 0; ch < 12; ++ch) {
      v16bf a = *(const v16bf*)&sBf[((mrow * 12 + ch) * 2 + half) * 16];
      v16bf b = *(const v16bf*)&img_pk[((size_t)(ch * HID + ncol) * 2 + half) * 16];
      acc = wmma_bf16(a, b, acc);
    }
#pragma unroll
    for (int r = 0; r < 8; ++r)
      sAcc[(half * 8 + r) * HID + ncol] = fmaxf(acc[r] + img_b[ncol], 0.f);   // imgS
  }
  // ---- lidar GEMM: K padded to 32, 1 WMMA ----
  {
    v8f acc = {0.f,0.f,0.f,0.f,0.f,0.f,0.f,0.f};
    v16bf a = *(const v16bf*)&sL[(mrow * 2 + half) * 16];
    v16bf b = *(const v16bf*)&lid_pk[((size_t)ncol * 2 + half) * 16];
    acc = wmma_bf16(a, b, acc);
#pragma unroll
    for (int r = 0; r < 8; ++r)
      sAcc[2048 + (half * 8 + r) * HID + ncol] = fmaxf(acc[r] + lidar_b[ncol], 0.f); // lidS
  }
  __syncthreads();

  // ---- concat [lidar | img] (16x256) -> swizzled bf16 in sBf ----
  for (int j = 0; j < 16; ++j) {
    int jj = cg * 16 + j;                       // 0..255
    float v = (jj < HID) ? sAcc[2048 + p * HID + jj] : sAcc[p * HID + (jj - HID)];
    sBf[swz_idx(p, 8, jj)] = (bf16_t)v;
  }
  __syncthreads();

  // ---- gate GEMM: (16x256)@(256x128): 8 WMMA ----
  {
    v8f acc = {0.f,0.f,0.f,0.f,0.f,0.f,0.f,0.f};
    for (int ch = 0; ch < 8; ++ch) {
      v16bf a = *(const v16bf*)&sBf[((mrow * 8 + ch) * 2 + half) * 16];
      v16bf b = *(const v16bf*)&gate_pk[((size_t)(ch * HID + ncol) * 2 + half) * 16];
      acc = wmma_bf16(a, b, acc);
    }
#pragma unroll
    for (int r = 0; r < 8; ++r)
      sH[(half * 8 + r) * HID + ncol] = fmaxf(acc[r] + gate_b1[ncol], 0.f);
  }
  __syncthreads();

  // ---- gate head (N=1), parallel partials + BEV index ----
  {
    float pd = 0.f;
    for (int j = 0; j < 8; ++j) {
      int k = cg * 8 + j;
      pd += sH[p * HID + k] * gate_w2[k];
    }
    sRed[0][p][cg] = pd;
  }
  if (t < 16) {
    int ix = (int)floorf((cx - XMINF) / RESF);
    int iy = (int)floorf((cy - YMINF) / RESF);
    int m = (ix >= 0) && (ix < WBEV) && (iy >= 0) && (iy < HBEV);
    sBevM[t]   = m;
    sBevIdx[t] = m ? (iy * WBEV + ix) : 0;
  }
  __syncthreads();
  if (cg == 0) {
    float d = gate_b2[0];
    for (int k = 0; k < 16; ++k) d += sRed[0][p][k];
    sGate[p] = 1.f / (1.f + expf(-d));
  }
  __syncthreads();

  // ---- fuse, store (contiguous per thread), scatter ----
  {
    float g  = sGate[p];
    int   bm = sBevM[p];
    int   bi = sBevIdx[p];
    int   c0 = cg * 8;
    float fbuf[8];
#pragma unroll
    for (int j = 0; j < 8; ++j) {
      float fi = sAcc[p * HID + c0 + j];          // img_emb
      float fl = sAcc[2048 + p * HID + c0 + j];   // lidar_emb
      fbuf[j]  = g * fi + (1.f - g) * fl;
    }
    float4* dst = (float4*)&fusedG[(size_t)(v0 + p) * HID + c0];
    dst[0] = make_float4(fbuf[0], fbuf[1], fbuf[2], fbuf[3]);
    dst[1] = make_float4(fbuf[4], fbuf[5], fbuf[6], fbuf[7]);
    if (bm) {
#pragma unroll
      for (int j = 0; j < 8; ++j) {
        atomicAdd(&bev_lid[(size_t)bi * HID + c0 + j], fbuf[j]);
        atomicAdd(&bev_img[(size_t)bi * HID + c0 + j], sAcc[p * HID + c0 + j]);
      }
    }
    if (t < 16 && bm) atomicAdd(&cnt[bi], 1.0f);
  }
}

// ===========================================================================
// Kernel B: BEV mean normalize
// ===========================================================================
__global__ void bev_norm_kernel(float* __restrict__ bimg, float* __restrict__ blid,
                                const float* __restrict__ cnt) {
  int cell = blockIdx.x;
  int c    = threadIdx.x;
  float d  = cnt[cell]; d = (d == 0.f) ? 1.f : d;
  float inv = 1.f / d;
  size_t o = (size_t)cell * HID + c;
  bimg[o] *= inv;
  blid[o] *= inv;
}

// ===========================================================================
// Kernel C/D: 3x3 conv as implicit GEMM (K = 9*128 = 1152 -> 36 chunks), WMMA
// block = 256 thr (8 waves); tile = 16-pixel x-strip x 128 out channels
// mode 0: out = relu(conv+b)   mode 1: out = addsrc + 0.5*(conv+b)  (in-place ok)
// Per chunk the tap is constant -> one bounds check + 4x float4 loads per lane.
// ===========================================================================
__global__ void __launch_bounds__(256)
conv3x3_kernel(const float* __restrict__ inp,      // NCELL x 128 (cell-major)
               const bf16_t* __restrict__ w_pk,    // packed B frags, 36 chunks
               const float* __restrict__ bias,
               const float* __restrict__ addsrc,
               float* __restrict__ outp,
               int mode)
{
  const int lane = threadIdx.x & 31;
  const int wid  = threadIdx.x >> 5;
  const int half = lane >> 4;
  const int mrow = lane & 15;
  const int ncol = wid * 16 + (lane & 15);
  const int x0   = blockIdx.x * 16;
  const int y    = blockIdx.y;
  const int x    = x0 + mrow;

  v8f acc = {0.f,0.f,0.f,0.f,0.f,0.f,0.f,0.f};
  for (int ch = 0; ch < 36; ++ch) {
    const int k0  = ch * 32;
    const int tap = k0 >> 7;           // constant within chunk (32 | 128)
    const int cb  = k0 & (HID - 1);
    const int yy  = y + tap / 3 - 1;
    const int xx  = x + tap % 3 - 1;
    v16bf a;
    if (xx >= 0 && xx < WBEV && yy >= 0 && yy < HBEV) {
      const float* src = inp + ((size_t)yy * WBEV + xx) * HID + cb + half * 8;
      float4 f0 = *(const float4*)(src);        // elements 0..3
      float4 f1 = *(const float4*)(src + 4);    // elements 4..7
      float4 f2 = *(const float4*)(src + 16);   // elements 8..11
      float4 f3 = *(const float4*)(src + 20);   // elements 12..15
      a[0]=(bf16_t)f0.x; a[1]=(bf16_t)f0.y; a[2]=(bf16_t)f0.z; a[3]=(bf16_t)f0.w;
      a[4]=(bf16_t)f1.x; a[5]=(bf16_t)f1.y; a[6]=(bf16_t)f1.z; a[7]=(bf16_t)f1.w;
      a[8]=(bf16_t)f2.x; a[9]=(bf16_t)f2.y; a[10]=(bf16_t)f2.z; a[11]=(bf16_t)f2.w;
      a[12]=(bf16_t)f3.x; a[13]=(bf16_t)f3.y; a[14]=(bf16_t)f3.z; a[15]=(bf16_t)f3.w;
    } else {
#pragma unroll
      for (int i = 0; i < 16; ++i) a[i] = (bf16_t)0.f;
    }
    v16bf b = *(const v16bf*)&w_pk[((size_t)(ch * HID + ncol) * 2 + half) * 16];
    acc = wmma_bf16(a, b, acc);
  }
#pragma unroll
  for (int r = 0; r < 8; ++r) {
    int px = x0 + half * 8 + r;
    if (px < WBEV) {
      size_t o = ((size_t)y * WBEV + px) * HID + ncol;
      float v = acc[r] + bias[ncol];
      outp[o] = (mode == 0) ? fmaxf(v, 0.f) : (addsrc[o] + 0.5f * v);
    }
  }
}

// ===========================================================================
// Kernel E: gather-back + LN + seg GEMM (128x32) -> out (V x 32, f32)
// block = 128 thr (4 waves; waves 0..1 do the WMMA)
// ===========================================================================
__global__ void __launch_bounds__(128)
final_seg_kernel(const float* __restrict__ fusedG, const float* __restrict__ coords,
                 const float* __restrict__ bevf,   const float* __restrict__ alphap,
                 const float* __restrict__ out_g,  const float* __restrict__ out_b,
                 const bf16_t* __restrict__ seg_pk, const float* __restrict__ seg_b,
                 float* __restrict__ out)
{
  __shared__ float  sP[16 * HID];
  __shared__ __attribute__((aligned(32))) bf16_t sA[16 * HID];
  __shared__ float  sRed[2][16][8];
  __shared__ float  sMean[16], sInv[16];
  __shared__ int    sIdx[16], sM[16];

  const int t  = threadIdx.x;
  const int v0 = blockIdx.x * 16;
  if (t < 16) {
    float x = coords[(size_t)(v0 + t) * 3 + 0];
    float y = coords[(size_t)(v0 + t) * 3 + 1];
    int ix = (int)floorf((x - XMINF) / RESF);
    int iy = (int)floorf((y - YMINF) / RESF);
    int m = (ix >= 0) && (ix < WBEV) && (iy >= 0) && (iy < HBEV);
    sM[t]   = m;
    sIdx[t] = m ? (iy * WBEV + ix) : 0;
  }
  __syncthreads();

  const float al = fminf(fmaxf(alphap[0], 0.f), 1.f);
  const int p  = t >> 3;    // 0..15
  const int g8 = t & 7;
  float ps = 0.f, pq = 0.f;
  for (int j = 0; j < 16; ++j) {
    int c = g8 * 16 + j;                             // contiguous 16 floats/thread
    float v = fusedG[(size_t)(v0 + p) * HID + c];
    if (sM[p]) v += al * bevf[(size_t)sIdx[p] * HID + c];
    sP[p * HID + c] = v;
    ps += v; pq += v * v;
  }
  sRed[0][p][g8] = ps; sRed[1][p][g8] = pq;
  __syncthreads();
  if (g8 == 0) {
    float s = 0.f, q = 0.f;
    for (int k = 0; k < 8; ++k) { s += sRed[0][p][k]; q += sRed[1][p][k]; }
    float mean = s * (1.0f / HID);
    float var  = fmaxf(q * (1.0f / HID) - mean * mean, 0.f);
    sMean[p] = mean; sInv[p] = rsqrtf(var + 1e-5f);
  }
  __syncthreads();
  for (int j = 0; j < 16; ++j) {
    int c = g8 * 16 + j;
    sA[swz_idx(p, 4, c)] =
        (bf16_t)((sP[p * HID + c] - sMean[p]) * sInv[p] * out_g[c] + out_b[c]);
  }
  __syncthreads();

  const int wid  = t >> 5;
  const int lane = t & 31;
  if (wid < 2) {                                     // wave-uniform; EXEC all-1 inside
    const int half = lane >> 4;
    const int mrow = lane & 15;
    const int ncol = wid * 16 + (lane & 15);         // 0..31
    v8f acc = {0.f,0.f,0.f,0.f,0.f,0.f,0.f,0.f};
    for (int ch = 0; ch < 4; ++ch) {
      v16bf a = *(const v16bf*)&sA[((mrow * 4 + ch) * 2 + half) * 16];
      v16bf b = *(const v16bf*)&seg_pk[((size_t)(ch * NCLS + ncol) * 2 + half) * 16];
      acc = wmma_bf16(a, b, acc);
    }
#pragma unroll
    for (int r = 0; r < 8; ++r) {
      int row = half * 8 + r;
      out[(size_t)(v0 + row) * NCLS + ncol] = acc[r] + seg_b[ncol];
    }
  }
}

// ===========================================================================
// Host launcher (graph-capture safe: only async ops on `stream`)
// ===========================================================================
extern "C" void kernel_launch(void* const* d_in, const int* in_sizes, int n_in,
                              void* d_out, int out_size, void* d_ws, size_t ws_size,
                              hipStream_t stream) {
  const float* pf         = (const float*)d_in[0];
  const float* imgfeat    = (const float*)d_in[1];
  const float* coords     = (const float*)d_in[2];
  const float* calib      = (const float*)d_in[3];
  const float* lidar_ln_g = (const float*)d_in[4];
  const float* lidar_ln_b = (const float*)d_in[5];
  const float* lidar_w    = (const float*)d_in[6];
  const float* lidar_b    = (const float*)d_in[7];
  const float* img_ln_g   = (const float*)d_in[8];
  const float* img_ln_b   = (const float*)d_in[9];
  const float* img_w      = (const float*)d_in[10];
  const float* img_b      = (const float*)d_in[11];
  const float* gate_w1    = (const float*)d_in[12];
  const float* gate_b1    = (const float*)d_in[13];
  const float* gate_w2    = (const float*)d_in[14];
  const float* gate_b2    = (const float*)d_in[15];
  const float* conv1_w    = (const float*)d_in[16];
  const float* conv1_b    = (const float*)d_in[17];
  const float* conv2_w    = (const float*)d_in[18];
  const float* conv2_b    = (const float*)d_in[19];
  const float* alpha      = (const float*)d_in[20];
  const float* out_ln_g   = (const float*)d_in[21];
  const float* out_ln_b   = (const float*)d_in[22];
  const float* seg_w      = (const float*)d_in[23];
  const float* seg_b      = (const float*)d_in[24];
  const int*   himg       = (const int*)d_in[25];
  const int*   wimg       = (const int*)d_in[26];

  // ---- workspace layout ----
  float* ws     = (float*)d_ws;
  float* fused  = ws;                                 // V*128 f32
  float* bevImg = fused  + (size_t)V_TOT * HID;       // NCELL*128
  float* bevLid = bevImg + (size_t)NCELL * HID;       // NCELL*128
  float* hb     = bevLid + (size_t)NCELL * HID;       // NCELL*128
  float* cnt    = hb     + (size_t)NCELL * HID;       // NCELL
  // packed bf16 weights (32B-aligned: offsets are multiples of 16 floats)
  size_t foff   = (size_t)NCELL + 16;  foff &= ~(size_t)15;
  bf16_t* pkBase = (bf16_t*)(cnt + foff);
  bf16_t* img_pk  = pkBase;                  // 12*128*32 = 49152
  bf16_t* gate_pk = img_pk  + 49152;         // 8*128*32  = 32768
  bf16_t* lid_pk  = gate_pk + 32768;         // 1*128*32  = 4096
  bf16_t* seg_pk  = lid_pk  + 4096;          // 4*32*32   = 4096
  bf16_t* c1_pk   = seg_pk  + 4096;          // 36*128*32 = 147456
  bf16_t* c2_pk   = c1_pk   + 147456;        // 147456

  // zero the scatter accumulators every call (bevImg & bevLid contiguous)
  hipMemsetAsync(bevImg, 0, (size_t)NCELL * HID * 2 * sizeof(float), stream);
  hipMemsetAsync(cnt,    0, (size_t)NCELL * sizeof(float), stream);

  // ---- weight packing (tiny) ----
  pack_weight_kernel<<<(49152 + 255) / 256, 256, 0, stream>>>(img_w,   img_pk,  DINO, HID, 49152);
  pack_weight_kernel<<<(32768 + 255) / 256, 256, 0, stream>>>(gate_w1, gate_pk, 256,  HID, 32768);
  pack_weight_kernel<<<(4096  + 255) / 256, 256, 0, stream>>>(lidar_w, lid_pk,  16,   HID, 4096);
  pack_weight_kernel<<<(4096  + 255) / 256, 256, 0, stream>>>(seg_w,   seg_pk,  HID,  NCLS, 4096);
  pack_conv_kernel  <<<147456 / 256,        256, 0, stream>>>(conv1_w, c1_pk);
  pack_conv_kernel  <<<147456 / 256,        256, 0, stream>>>(conv2_w, c2_pk);

  const int ntile = V_TOT / 16;   // 9375, exact

  fuse_point_kernel<<<ntile, 256, 0, stream>>>(
      pf, imgfeat, coords, calib,
      lidar_ln_g, lidar_ln_b, lid_pk, lidar_b,
      img_ln_g, img_ln_b, img_pk, img_b,
      gate_pk, gate_b1, gate_w2, gate_b2,
      himg, wimg,
      fused, bevImg, bevLid, cnt);

  bev_norm_kernel<<<NCELL, HID, 0, stream>>>(bevImg, bevLid, cnt);

  dim3 cgrid((WBEV + 15) / 16, HBEV);
  conv3x3_kernel<<<cgrid, 256, 0, stream>>>(bevImg, c1_pk, conv1_b, nullptr, hb, 0);
  conv3x3_kernel<<<cgrid, 256, 0, stream>>>(hb, c2_pk, conv2_b, bevLid, bevLid, 1);

  final_seg_kernel<<<ntile, 128, 0, stream>>>(
      fused, coords, bevLid, alpha, out_ln_g, out_ln_b, seg_pk, seg_b, (float*)d_out);
}
